// LocalAttention_16432544874779
// MI455X (gfx1250) — compile-verified, matrix operand order checked by reading, not run on hardware
//
#include <hip/hip_runtime.h>
#include <hip/hip_bf16.h>

typedef __attribute__((ext_vector_type(16))) _Float16 v16h;
typedef __attribute__((ext_vector_type(8)))  _Float16 v8h;
typedef __attribute__((ext_vector_type(8)))  float    v8f;
typedef __attribute__((ext_vector_type(4)))  int      v4i;

#define B_  4
#define N_  8192
#define DIM_ 1024
#define H_  16
#define DH_ 64
#define W_  128
#define BN_ (B_ * N_)          // 32768 rows
#define INNER_ 1024

#define AS1 __attribute__((address_space(1)))
#define AS3 __attribute__((address_space(3)))

#if __has_builtin(__builtin_amdgcn_global_load_async_to_lds_b128)
#define HAVE_ASYNC_LDS 1
#endif

// 16-byte global -> LDS copy; async (ASYNCcnt-tracked) when available.
static __device__ __forceinline__ void copy16_g2l(const _Float16* g, _Float16* l) {
#ifdef HAVE_ASYNC_LDS
    __builtin_amdgcn_global_load_async_to_lds_b128(
        (AS1 v4i*)g, (AS3 v4i*)l, /*offset=*/0, /*cpol=*/0);
#else
    *(v8h*)l = *(const v8h*)g;
#endif
}

static __device__ __forceinline__ void async_copies_wait() {
#ifdef HAVE_ASYNC_LDS
  #if __has_builtin(__builtin_amdgcn_s_wait_asynccnt)
    __builtin_amdgcn_s_wait_asynccnt(0);
  #else
    asm volatile("s_wait_asynccnt 0" ::: "memory");
  #endif
#endif
}

static __device__ __forceinline__ v8f wmma_f16(v16h a, v16h b, v8f c) {
    return __builtin_amdgcn_wmma_f32_16x16x32_f16(
        /*neg_a=*/false, a, /*neg_b=*/false, b,
        /*c_mod=*/(short)0, c, /*reuse_a=*/false, /*reuse_b=*/false);
}

// Load a 16x32 f16 A-layout fragment from an LDS tile (row-major, stride `ld` halves).
// lane<16: row M=lane, K {0..7, 16..23};  lane>=16: row M=lane-16, K {8..15, 24..31}.
static __device__ __forceinline__ v16h load_frag(const _Float16* base, int ld, int lane) {
    const int m  = lane & 15;
    const int ko = (lane >> 4) << 3;        // 0 or 8
    union { v16h v; v8h h[2]; } u;
    u.h[0] = *(const v8h*)(base + m * ld + ko);
    u.h[1] = *(const v8h*)(base + m * ld + ko + 16);
    return u.v;
}

// ---------------------------------------------------------------------------
// Tiled GEMM: C[M,N] = A[M,K] * Bw[K,N]  (A fp32 or f16; C f16*alpha or fp32+bias)
// Block tile 128x128, 8 waves (2x4), wave tile 64x32, k-step 32.
// Double-buffered LDS: fetch tile k+1 to registers while WMMAs consume tile k.
// ---------------------------------------------------------------------------
template<bool AF16, bool F16OUT>
__global__ __launch_bounds__(256) void gemm_kernel(
    const float* __restrict__ Af, const _Float16* __restrict__ Ah,
    const float* __restrict__ Bw,
    _Float16* __restrict__ Cf16, float* __restrict__ Cf32,
    const float* __restrict__ bias, float alpha, int M, int N, int K)
{
    constexpr int LDT = 40;                      // 80B rows: 16B-aligned, bank-spread
    __shared__ _Float16 sA[2][128 * LDT];
    __shared__ _Float16 sB[2][128 * LDT];        // stores Bw^T tile: [n][k]

    const int t    = threadIdx.x;
    const int lane = t & 31;
    const int wave = t >> 5;
    const int wm   = wave >> 2;                  // 0..1
    const int wn   = wave & 3;                   // 0..3
    const int row0 = blockIdx.y * 128;
    const int col0 = blockIdx.x * 128;

    auto fetchA = [&](int kk, v8h* av) {
        #pragma unroll
        for (int cc = 0; cc < 2; ++cc) {
            int c  = t + 256 * cc;               // 0..511
            int i  = c >> 2;                     // row 0..127
            int k8 = (c & 3) << 3;               // 0,8,16,24
            if constexpr (AF16) {
                av[cc] = *(const v8h*)(Ah + (size_t)(row0 + i) * K + kk + k8);
            } else {
                const float* ap = Af + (size_t)(row0 + i) * K + kk + k8;
                #pragma unroll
                for (int e = 0; e < 8; ++e) av[cc][e] = (_Float16)ap[e];
            }
        }
    };
    auto fetchB = [&](int kk, _Float16* bvv) {
        #pragma unroll
        for (int ii = 0; ii < 16; ++ii) {
            int e  = t + 256 * ii;               // 0..4095
            int kb = e >> 7;                     // 0..31
            int n  = e & 127;
            bvv[ii] = (_Float16)Bw[(size_t)(kk + kb) * N + col0 + n];
        }
    };
    auto storeAB = [&](int buf, const v8h* av, const _Float16* bvv) {
        #pragma unroll
        for (int cc = 0; cc < 2; ++cc) {
            int c  = t + 256 * cc;
            int i  = c >> 2;
            int k8 = (c & 3) << 3;
            *(v8h*)(sA[buf] + i * LDT + k8) = av[cc];
        }
        #pragma unroll
        for (int ii = 0; ii < 16; ++ii) {
            int e  = t + 256 * ii;
            int kb = e >> 7;
            int n  = e & 127;
            sB[buf][n * LDT + kb] = bvv[ii];
        }
    };

    v8f acc[4][2];
    #pragma unroll
    for (int i = 0; i < 4; ++i)
        #pragma unroll
        for (int j = 0; j < 2; ++j) { v8f z = {}; acc[i][j] = z; }

    v8h av[2];
    _Float16 bv[16];
    fetchA(0, av);
    fetchB(0, bv);
    storeAB(0, av, bv);

    int buf = 0;
    for (int kk = 0; kk < K; kk += 32) {
        __syncthreads();                         // LDS[buf] ready for all waves
        const bool more = (kk + 32) < K;
        if (more) { fetchA(kk + 32, av); fetchB(kk + 32, bv); }   // overlap with WMMAs

        v16h afrag[4], bfrag[2];
        #pragma unroll
        for (int mi = 0; mi < 4; ++mi)
            afrag[mi] = load_frag(sA[buf] + (wm * 64 + mi * 16) * LDT, LDT, lane);
        #pragma unroll
        for (int ni = 0; ni < 2; ++ni)
            bfrag[ni] = load_frag(sB[buf] + (wn * 32 + ni * 16) * LDT, LDT, lane);
        #pragma unroll
        for (int mi = 0; mi < 4; ++mi)
            #pragma unroll
            for (int ni = 0; ni < 2; ++ni)
                acc[mi][ni] = wmma_f16(afrag[mi], bfrag[ni], acc[mi][ni]);

        if (more) storeAB(buf ^ 1, av, bv);      // safe: buf^1 last read in prev iter, before this iter's barrier
        buf ^= 1;
    }

    // ---- epilogue: C layout lane l -> row r + 8*(l>>4), col l&15 ----
    const int rb = (lane >> 4) << 3;
    const int cb = lane & 15;
    #pragma unroll
    for (int mi = 0; mi < 4; ++mi) {
        #pragma unroll
        for (int ni = 0; ni < 2; ++ni) {
            int rbase = row0 + wm * 64 + mi * 16 + rb;
            int c     = col0 + wn * 32 + ni * 16 + cb;
            #pragma unroll
            for (int r = 0; r < 8; ++r) {
                float vl = acc[mi][ni][r] * alpha;
                size_t off = (size_t)(rbase + r) * N + c;
                if constexpr (F16OUT) Cf16[off] = (_Float16)vl;
                else                  Cf32[off] = vl + bias[c];
            }
        }
    }
}

// ---------------------------------------------------------------------------
// Windowed causal attention, one block per (head, window).
// Q 128x64 (pre-scaled), K/V 256x64 = [prev window | cur window] (window 0: zeros).
// 8 waves; each wave owns a 16-row query strip; S kept in registers.
// Q/K staged with async global->LDS copies when the toolchain exposes them.
// ---------------------------------------------------------------------------
__global__ __launch_bounds__(256) void attn_kernel(
    const _Float16* __restrict__ q,    // (BN, 1024), already * dh^-0.5
    const _Float16* __restrict__ kv,   // (BN, 2048): [:,0:1024]=K, [:,1024:2048]=V
    _Float16* __restrict__ o)          // (BN, 1024)
{
    constexpr int LDQK = 88;           // 176B rows (16B-aligned)
    constexpr int LDVP = 264;          // 528B rows (16B-aligned)
    // Q (128*88=11264) | K (256*88=22528) | Vt (64*264=16896); P (128*264=33792) reuses Q+K.
    __shared__ _Float16 smem[50688];
    _Float16* sQ  = smem;
    _Float16* sK  = smem + 11264;
    _Float16* sVt = smem + 33792;
    _Float16* sP  = smem;              // overlaps Q+K exactly

    const int t    = threadIdx.x;
    const int lane = t & 31;
    const int wave = t >> 5;
    const int w0   = blockIdx.x;       // window 0..63
    const int bh   = blockIdx.y;       // 0..63
    const int b    = bh >> 4;
    const int h    = bh & 15;
    const size_t rowQ0 = (size_t)b * N_ + (size_t)w0 * W_;
    const int hcol = h * DH_;

    // ---- stage Q (128x64) ----
    #pragma unroll
    for (int ii = 0; ii < 4; ++ii) {
        int c = t + 256 * ii;          // 0..1023 chunks of 8 halves
        int i = c >> 3, d8 = (c & 7) << 3;
        copy16_g2l(q + (rowQ0 + i) * 1024 + hcol + d8, sQ + i * LDQK + d8);
    }
    // ---- stage K (256x64): rows 0..127 = prev window (zero for w0==0) ----
    #pragma unroll
    for (int ii = 0; ii < 8; ++ii) {
        int c = t + 256 * ii;          // 0..2047
        int j = c >> 3, d8 = (c & 7) << 3;
        if (w0 == 0 && j < 128) {
            v8h z = {};
            *(v8h*)(sK + j * LDQK + d8) = z;
        } else {
            size_t rowg = rowQ0 - 128 + j;
            copy16_g2l(kv + rowg * 2048 + hcol + d8, sK + j * LDQK + d8);
        }
    }
    // ---- stage V transposed: sVt[d][j] (manual transpose; not async-able) ----
    #pragma unroll
    for (int ii = 0; ii < 8; ++ii) {
        int c = t + 256 * ii;
        int j = c >> 3, d8 = (c & 7) << 3;
        v8h hv = {};
        if (!(w0 == 0 && j < 128)) {
            size_t rowg = rowQ0 - 128 + j;
            hv = *(const v8h*)(kv + rowg * 2048 + 1024 + hcol + d8);
        }
        #pragma unroll
        for (int e = 0; e < 8; ++e) sVt[(d8 + e) * LDVP + j] = hv[e];
    }
    async_copies_wait();
    __syncthreads();

    // ---- S = Q K^T for this wave's 16-row strip ----
    v16h qf0 = load_frag(sQ + wave * 16 * LDQK,      LDQK, lane);
    v16h qf1 = load_frag(sQ + wave * 16 * LDQK + 32, LDQK, lane);
    v8f s[16];
    #pragma unroll
    for (int jb = 0; jb < 16; ++jb) {
        v8f a = {};
        v16h kf0 = load_frag(sK + jb * 16 * LDQK,      LDQK, lane);
        v16h kf1 = load_frag(sK + jb * 16 * LDQK + 32, LDQK, lane);
        a = wmma_f16(qf0, kf0, a);
        a = wmma_f16(qf1, kf1, a);
        s[jb] = a;
    }
    __syncthreads();   // all waves done reading sQ/sK; sP may now overwrite them

    // ---- masked softmax in registers; write P (f16) to LDS ----
    const int cb = lane & 15;
    const int rb = (lane >> 4) << 3;
    #pragma unroll
    for (int r = 0; r < 8; ++r) {
        const int i = wave * 16 + rb + r;          // query row in [0,128)
        float m = -3.0e38f;
        #pragma unroll
        for (int jb = 0; jb < 16; ++jb) {
            int col = jb * 16 + cb;                // key col in [0,256)
            float val = (col > i + 128) ? -3.0e38f : s[jb][r];
            s[jb][r] = val;
            m = fmaxf(m, val);
        }
        #pragma unroll
        for (int off = 1; off < 16; off <<= 1) m = fmaxf(m, __shfl_xor(m, off, 32));
        float sum = 0.f;
        #pragma unroll
        for (int jb = 0; jb < 16; ++jb) {
            float p = __expf(s[jb][r] - m);
            s[jb][r] = p;
            sum += p;
        }
        #pragma unroll
        for (int off = 1; off < 16; off <<= 1) sum += __shfl_xor(sum, off, 32);
        float inv = 1.0f / sum;
        #pragma unroll
        for (int jb = 0; jb < 16; ++jb)
            sP[(wave * 16 + rb + r) * LDVP + jb * 16 + cb] = (_Float16)(s[jb][r] * inv);
    }
    __syncthreads();

    // ---- O = P V : 16x64 per wave, K-dim 256 in 8 steps ----
    v8f oacc[4];
    #pragma unroll
    for (int ni = 0; ni < 4; ++ni) { v8f z = {}; oacc[ni] = z; }
    #pragma unroll
    for (int kb = 0; kb < 8; ++kb) {
        v16h pf = load_frag(sP + wave * 16 * LDVP + kb * 32, LDVP, lane);
        #pragma unroll
        for (int ni = 0; ni < 4; ++ni) {
            v16h vf = load_frag(sVt + ni * 16 * LDVP + kb * 32, LDVP, lane);
            oacc[ni] = wmma_f16(pf, vf, oacc[ni]);
        }
    }
    // ---- store O (f16) ----
    #pragma unroll
    for (int ni = 0; ni < 4; ++ni) {
        int cg = hcol + ni * 16 + cb;
        #pragma unroll
        for (int r = 0; r < 8; ++r) {
            size_t row = rowQ0 + wave * 16 + rb + r;
            o[row * 1024 + cg] = (_Float16)oacc[ni][r];
        }
    }
}

extern "C" void kernel_launch(void* const* d_in, const int* in_sizes, int n_in,
                              void* d_out, int out_size, void* d_ws, size_t ws_size,
                              hipStream_t stream) {
    const float* x   = (const float*)d_in[0];   // (4, 8192, 1024)
    const float* wq  = (const float*)d_in[1];   // (1024, 1024)
    const float* wkv = (const float*)d_in[2];   // (1024, 2048)
    const float* wo  = (const float*)d_in[3];   // (1024, 1024)
    const float* bo  = (const float*)d_in[4];   // (1024,)
    float* out = (float*)d_out;                 // (4, 8192, 1024) fp32

    _Float16* qh  = (_Float16*)d_ws;                         // 32768*1024 f16 (pre-scaled Q)
    _Float16* kvh = qh  + (size_t)BN_ * INNER_;              // 32768*2048 f16
    _Float16* aoh = kvh + (size_t)BN_ * 2 * INNER_;          // 32768*1024 f16

    const float scale = 0.125f;   // dh^-0.5, folded into Q projection

    // Q = (X @ Wq) * scale  -> f16
    gemm_kernel<false, true><<<dim3(INNER_ / 128, BN_ / 128), 256, 0, stream>>>(
        x, (const _Float16*)nullptr, wq, qh, (float*)nullptr,
        (const float*)nullptr, scale, BN_, INNER_, DIM_);

    // KV = X @ Wkv -> f16
    gemm_kernel<false, true><<<dim3(2 * INNER_ / 128, BN_ / 128), 256, 0, stream>>>(
        x, (const _Float16*)nullptr, wkv, kvh, (float*)nullptr,
        (const float*)nullptr, 1.0f, BN_, 2 * INNER_, DIM_);

    // windowed attention -> f16
    attn_kernel<<<dim3(N_ / W_, B_ * H_), 256, 0, stream>>>(qh, kvh, aoh);

    // out = AO @ Wo + bo -> fp32
    gemm_kernel<true, false><<<dim3(DIM_ / 128, BN_ / 128), 256, 0, stream>>>(
        (const float*)nullptr, aoh, wo, (_Float16*)nullptr, out,
        bo, 1.0f, BN_, DIM_, INNER_);
}
